// MultiHeadDotProductAttention_74380243632266
// MI455X (gfx1250) — compile-verified, matrix operand order checked by reading, not run on hardware
//
#include <hip/hip_runtime.h>
#include <hip/hip_bf16.h>

// ---------------------------------------------------------------------------
// Flash-attention for MI455X (gfx1250, wave32, WMMA 16x16x32 f16).
// query/key/value: [B=2, S=2048, H=16, D=128] f32
// bias:            [1, 1, S, S] f32 (causal 0/-1e9)
// q_scale/k_scale: [D] f32
// keep_mask:       [B, H, 1, S] bool (1 byte each)
// out:             [B, S, H, D] f32
// ---------------------------------------------------------------------------

typedef __attribute__((ext_vector_type(16))) _Float16 v16h;
typedef __attribute__((ext_vector_type(8)))  _Float16 v8h;
typedef __attribute__((ext_vector_type(8)))  float    v8f;

#define NB   2
#define NS   2048
#define NH   16
#define ND   128
#define QT   16      // q rows per wave
#define NW   4       // waves per workgroup
#define WGQ  (QT*NW) // 64 q rows per workgroup
#define KT   32      // keys per iteration
#define VPAD 40      // padded vbuf row (halves); 80B -> 16B aligned rows

static __device__ __forceinline__ v16h cat16(v8h a, v8h b) {
  return __builtin_shufflevector(a, b, 0,1,2,3,4,5,6,7,8,9,10,11,12,13,14,15);
}

// Load a 16-half A/B fragment: halves 0..7 from base+off_lo, 8..15 from base+off_hi.
// Both offsets are 16B aligned by construction -> ds_load_b128 pairs.
static __device__ __forceinline__ v16h ldfrag(const _Float16* base, int off_lo, int off_hi) {
  v8h lo = *(const v8h*)(base + off_lo);
  v8h hi = *(const v8h*)(base + off_hi);
  return cat16(lo, hi);
}

__global__ __launch_bounds__(NW * 32)
void mhsa_flash_wmma_kernel(const float* __restrict__ Q,
                            const float* __restrict__ K,
                            const float* __restrict__ V,
                            const float* __restrict__ Bias,
                            const float* __restrict__ Qs,
                            const float* __restrict__ Ks,
                            const unsigned char* __restrict__ Keep,
                            float* __restrict__ Out) {
  __shared__ _Float16 kbuf[KT][ND];        // RMSNormed keys, row-major (f16)
  __shared__ _Float16 vbuf[ND][VPAD];      // V transposed: vbuf[d][k]
  __shared__ _Float16 qbuf[NW][QT][ND];    // RMSNormed q tiles (f16)
  __shared__ _Float16 pbuf[NW][QT][KT];    // per-wave P staging (C-layout -> A-layout)

  const int tid  = threadIdx.x;
  const int lane = tid & 31;
  const int w    = tid >> 5;
  const int g    = lane >> 4;      // lane group (bit 4)
  const int ln   = lane & 15;

  const int bid  = blockIdx.x;
  const int qblk = bid & (NS / WGQ - 1);   // 32 q-blocks
  const int h    = (bid >> 5) & (NH - 1);
  const int b    = bid >> 9;
  const int q0   = qblk * WGQ;
  const int wq0  = q0 + w * QT;

  const size_t bh_row   = ((size_t)b * NS) * NH + h;   // row stride unit = NH*ND
  const float* Qbase    = Q + bh_row * ND;             // + s*NH*ND
  const float* Kbase    = K + bh_row * ND;
  const float* Vbase    = V + bh_row * ND;
  const unsigned char* km = Keep + ((size_t)b * NH + h) * NS;

  // ---------------- Q: load, RMSNorm, f16 -> qbuf (2 lanes per row) ---------
  {
    const int row  = lane >> 1;
    const int half = lane & 1;
    const float* src = Qbase + (size_t)(wq0 + row) * NH * ND + half * 64;
    float4 t[16];
    float ss = 0.f;
#pragma unroll
    for (int i = 0; i < 16; ++i) {
      t[i] = ((const float4*)src)[i];
      ss += t[i].x*t[i].x + t[i].y*t[i].y + t[i].z*t[i].z + t[i].w*t[i].w;
    }
    ss += __shfl_xor(ss, 1, 32);                       // combine the two halves
    const float inv = __frsqrt_rn(ss * (1.0f / ND) + 1e-6f);
#pragma unroll
    for (int i = 0; i < 16; ++i) {
      const int d0 = half * 64 + i * 4;
      qbuf[w][row][d0+0] = (_Float16)(t[i].x * inv * Qs[d0+0]);
      qbuf[w][row][d0+1] = (_Float16)(t[i].y * inv * Qs[d0+1]);
      qbuf[w][row][d0+2] = (_Float16)(t[i].z * inv * Qs[d0+2]);
      qbuf[w][row][d0+3] = (_Float16)(t[i].w * inv * Qs[d0+3]);
    }
  }

  // A-layout Q fragments (same-wave DS ops are in-order; no barrier needed).
  // lane ln holds row M=ln; halves 0..7 -> Kd = 32c + 8g .. +7, 8..15 -> +16.
  v16h qa[4];
#pragma unroll
  for (int c = 0; c < 4; ++c)
    qa[c] = ldfrag(&qbuf[w][ln][0], c*32 + g*8, c*32 + 16 + g*8);

  v8f acc[8] = {};                     // O accumulator: 16 x 128 (8 N-tiles)
  float mrow[8], lrow[8];
#pragma unroll
  for (int r = 0; r < 8; ++r) { mrow[r] = -1e30f; lrow[r] = 0.f; }

  const int niter = (q0 + WGQ) / KT;   // causal: keys <= q0+63

  for (int it = 0; it < niter; ++it) {
    const int kb0 = it * KT;
    __syncthreads();                   // previous-iteration readers done

    // ---- cooperative K RMSNorm + V transpose into LDS (4 threads / row) ----
    {
      const int row  = tid >> 2;       // 0..31 keys
      const int part = tid & 3;        // 32 d-elements each
      const float* src = Kbase + (size_t)(kb0 + row) * NH * ND + part * 32;
      float4 t[8];
      float ss = 0.f;
#pragma unroll
      for (int i = 0; i < 8; ++i) {
        t[i] = ((const float4*)src)[i];
        ss += t[i].x*t[i].x + t[i].y*t[i].y + t[i].z*t[i].z + t[i].w*t[i].w;
      }
      ss += __shfl_xor(ss, 1, 32);
      ss += __shfl_xor(ss, 2, 32);     // quad (parts 0..3) reduction
      const float inv = __frsqrt_rn(ss * (1.0f / ND) + 1e-6f);
#pragma unroll
      for (int i = 0; i < 8; ++i) {
        const int d0 = part * 32 + i * 4;
        kbuf[row][d0+0] = (_Float16)(t[i].x * inv * Ks[d0+0]);
        kbuf[row][d0+1] = (_Float16)(t[i].y * inv * Ks[d0+1]);
        kbuf[row][d0+2] = (_Float16)(t[i].z * inv * Ks[d0+2]);
        kbuf[row][d0+3] = (_Float16)(t[i].w * inv * Ks[d0+3]);
      }
      const float* vsrc = Vbase + (size_t)(kb0 + row) * NH * ND + part * 32;
#pragma unroll
      for (int i = 0; i < 8; ++i) {
        float4 tv = ((const float4*)vsrc)[i];
        const int d0 = part * 32 + i * 4;
        vbuf[d0+0][row] = (_Float16)tv.x;
        vbuf[d0+1][row] = (_Float16)tv.y;
        vbuf[d0+2][row] = (_Float16)tv.z;
        vbuf[d0+3][row] = (_Float16)tv.w;
      }
      if (it + 1 < niter) {            // CDNA5 prefetch of next tile
        const float* nk = Kbase + (size_t)(kb0 + KT + row) * NH * ND + part * 32;
        const float* nv = Vbase + (size_t)(kb0 + KT + row) * NH * ND + part * 32;
        __builtin_prefetch(nk, 0, 1);
        __builtin_prefetch(nv, 0, 1);
      }
    }
    __syncthreads();

    // ---- S = Q Kt : two 16x16 score tiles (keys 0-15 / 16-31) -------------
    v8f st0 = {}, st1 = {};
#pragma unroll
    for (int c = 0; c < 4; ++c) {
      v16h b0 = ldfrag(&kbuf[ln][0],      c*32 + g*8, c*32 + 16 + g*8);
      v16h b1 = ldfrag(&kbuf[16 + ln][0], c*32 + g*8, c*32 + 16 + g*8);
      st0 = __builtin_amdgcn_wmma_f32_16x16x32_f16(false, qa[c], false, b0,
                                                   (short)0, st0, false, false);
      st1 = __builtin_amdgcn_wmma_f32_16x16x32_f16(false, qa[c], false, b1,
                                                   (short)0, st1, false, false);
    }

    // ---- bias + online softmax (row stats live along the C-layout rows) ---
    const float dmul0 = km[kb0 + ln]      ? (1.0f / 0.9f) : 0.0f;
    const float dmul1 = km[kb0 + 16 + ln] ? (1.0f / 0.9f) : 0.0f;
    float scale[8], p0[8], p1[8];
#pragma unroll
    for (int r = 0; r < 8; ++r) {
      const int qrow = wq0 + r + 8 * g;
      const float* bp = Bias + (size_t)qrow * NS + kb0;
      const float s0 = st0[r] + bp[ln];
      const float s1 = st1[r] + bp[16 + ln];
      float mt = fmaxf(s0, s1);
      mt = fmaxf(mt, __shfl_xor(mt, 1, 32));
      mt = fmaxf(mt, __shfl_xor(mt, 2, 32));
      mt = fmaxf(mt, __shfl_xor(mt, 4, 32));
      mt = fmaxf(mt, __shfl_xor(mt, 8, 32));          // stays in 16-lane half
      const float mn = fmaxf(mrow[r], mt);
      scale[r] = __expf(mrow[r] - mn);
      const float e0 = __expf(s0 - mn);
      const float e1 = __expf(s1 - mn);
      float rs = e0 + e1;
      rs += __shfl_xor(rs, 1, 32);
      rs += __shfl_xor(rs, 2, 32);
      rs += __shfl_xor(rs, 4, 32);
      rs += __shfl_xor(rs, 8, 32);
      lrow[r] = lrow[r] * scale[r] + rs;              // denominator pre-dropout
      mrow[r] = mn;
      p0[r] = e0; p1[r] = e1;
    }
#pragma unroll
    for (int t = 0; t < 8; ++t)
#pragma unroll
      for (int r = 0; r < 8; ++r) acc[t][r] *= scale[r];

    // ---- P (with dropout) -> LDS, reload in A layout (in-wave transpose) --
#pragma unroll
    for (int r = 0; r < 8; ++r) {
      pbuf[w][r + 8*g][ln]      = (_Float16)(p0[r] * dmul0);
      pbuf[w][r + 8*g][16 + ln] = (_Float16)(p1[r] * dmul1);
    }
    const v16h pa = ldfrag(&pbuf[w][ln][0], g*8, 16 + g*8);

    // ---- O += P V : 8 column tiles of D ----------------------------------
#pragma unroll
    for (int t = 0; t < 8; ++t) {
      const v16h vb = ldfrag(&vbuf[t*16 + ln][0], g*8, 16 + g*8);
      acc[t] = __builtin_amdgcn_wmma_f32_16x16x32_f16(false, pa, false, vb,
                                                      (short)0, acc[t], false, false);
    }
  }

  // ---------------- epilogue: normalize and store --------------------------
  float linv[8];
#pragma unroll
  for (int r = 0; r < 8; ++r) linv[r] = 1.0f / lrow[r];
#pragma unroll
  for (int t = 0; t < 8; ++t)
#pragma unroll
    for (int r = 0; r < 8; ++r) {
      const int qrow = wq0 + r + 8 * g;
      Out[(((size_t)b * NS + qrow) * NH + h) * ND + t * 16 + ln] = acc[t][r] * linv[r];
    }
}

extern "C" void kernel_launch(void* const* d_in, const int* in_sizes, int n_in,
                              void* d_out, int out_size, void* d_ws, size_t ws_size,
                              hipStream_t stream) {
  (void)in_sizes; (void)n_in; (void)out_size; (void)d_ws; (void)ws_size;
  const float* q    = (const float*)d_in[0];
  const float* k    = (const float*)d_in[1];
  const float* v    = (const float*)d_in[2];
  const float* bias = (const float*)d_in[3];
  const float* qs   = (const float*)d_in[4];
  const float* ks   = (const float*)d_in[5];
  const unsigned char* keep = (const unsigned char*)d_in[6];  // numpy bool = 1 byte
  float* out = (float*)d_out;

  const int blocks = NB * NH * (NS / WGQ);   // 2*16*32 = 1024
  mhsa_flash_wmma_kernel<<<blocks, NW * 32, 0, stream>>>(q, k, v, bias, qs, ks, keep, out);
}
